// UniMatchRNN_25744033973062
// MI455X (gfx1250) — matrix operand order repaired and checked by reading, not compiled, not run on hardware
//
#include <hip/hip_runtime.h>
#include <hip/hip_bf16.h>

#define B_   128
#define P_   512
#define Q_   128
#define DP_  512
#define DQ_  512
#define H_   512
#define R_   1024
#define G4_  2048      // 4*H
#define KD_  1536      // R + H  (fused [zi|h] K-dim)
#define NWG  64
#define NT   256
#define LN_EPS 1e-5f

typedef __attribute__((ext_vector_type(16))) __bf16 v16bf;
typedef __attribute__((ext_vector_type(8)))  float  v8f;

// ---------------- workspace layout (bytes) ----------------
static constexpr size_t OFF_BAR = 0;                              // 2 u32 (pad 256)
static constexpr size_t OFF_H0  = 256;                            // B*H f32 (h ping)
static constexpr size_t OFF_H1  = OFF_H0  + (size_t)B_*H_*4;      // B*H f32 (h pong)
static constexpr size_t OFF_C   = OFF_H1  + (size_t)B_*H_*4;
static constexpr size_t OFF_PRE = OFF_C   + (size_t)B_*H_*4;
static constexpr size_t OFF_QA  = OFF_PRE + (size_t)B_*H_*4;
static constexpr size_t OFF_ZI  = OFF_QA  + (size_t)B_*DQ_*4;
static constexpr size_t OFF_NEG = OFF_ZI  + (size_t)B_*R_*4;
static constexpr size_t OFF_WQH = OFF_NEG + (size_t)B_*Q_*4;
static constexpr size_t OFF_WPR = OFF_WQH + (size_t)B_*Q_*H_*4;   // wq_hq (B*Q, H) f32
static constexpr size_t OFF_WQT = OFF_WPR + (size_t)H_*1024*2;    // WprT (512,1024) bf16
static constexpr size_t OFF_WGT = OFF_WQT + (size_t)H_*DQ_*2;     // WqT (512,512) bf16
static constexpr size_t OFF_WIH = OFF_WGT + (size_t)R_*R_*2;      // WgateT (1024,1024) bf16
// Wihh (2048,1536) bf16 follows

__device__ __forceinline__ float fast_tanh(float x) {
#if __has_builtin(__builtin_amdgcn_tanhf)
  return __builtin_amdgcn_tanhf(x);          // gfx1250 v_tanh_f32
#else
  return tanhf(x);
#endif
}

__device__ __forceinline__ float sigm(float x) {
  return __builtin_amdgcn_rcpf(1.0f + __expf(-x));
}

__device__ __forceinline__ v8f wmma_bf16(v16bf a, v16bf b, v8f c) {
  return __builtin_amdgcn_wmma_f32_16x16x32_bf16(false, a, false, b, (short)0, c, false, false);
}

// B fragment: Bt is N x K row-major bf16 (original W transposed).
// lane n=lane&15 holds column col0+n, K-values [k0+16*hi, +16) contiguous.
__device__ __forceinline__ v16bf load_b_frag(const __bf16* Bt, int ldb, int col0, int k0, int lane) {
  int n = lane & 15, hi = lane >> 4;
  const __bf16* p = Bt + (size_t)(col0 + n) * ldb + (k0 + 16 * hi);
  return *(const v16bf*)p;  // 32B aligned
}

// A fragment from fp32, converted to bf16 on the fly. Logical A row r, K index k:
// k < kb -> p0 (row stride s0), else p1 (row stride s1, index k-kb).
__device__ __forceinline__ v16bf load_a_frag2(const float* p0, int s0, int kb,
                                              const float* p1, int s1,
                                              int row0, int k0, int lane) {
  int m = lane & 15, hi = lane >> 4;
  int r = row0 + m;
  v16bf a;
#pragma unroll
  for (int cc = 0; cc < 2; ++cc) {
    int kc = k0 + 16 * hi + 8 * cc;
    const float* src = (kc < kb) ? (p0 + (size_t)r * s0 + kc)
                                 : (p1 + (size_t)r * s1 + (kc - kb));
    float4 x = ((const float4*)src)[0];
    float4 y = ((const float4*)src)[1];
    int o = cc * 8;
    a[o + 0] = (__bf16)x.x; a[o + 1] = (__bf16)x.y; a[o + 2] = (__bf16)x.z; a[o + 3] = (__bf16)x.w;
    a[o + 4] = (__bf16)y.x; a[o + 5] = (__bf16)y.y; a[o + 6] = (__bf16)y.z; a[o + 7] = (__bf16)y.w;
  }
  return a;
}

// device-wide sense-reversing barrier (grid must be co-resident: 64 WGs)
__device__ __forceinline__ void grid_barrier(unsigned* bar) {
  __syncthreads();
  if (threadIdx.x == 0) {
    __threadfence();
    unsigned gen = atomicAdd(&bar[1], 0u);
    unsigned prev = atomicAdd(&bar[0], 1u);
    if (prev == (unsigned)(gridDim.x - 1)) {
      atomicExch(&bar[0], 0u);
      __threadfence();
      atomicAdd(&bar[1], 1u);
    } else {
      while (atomicAdd(&bar[1], 0u) == gen) { __builtin_amdgcn_s_sleep(2); }
    }
    __threadfence();
  }
  __syncthreads();
}

// ---------------- one-time: transpose+convert weights to bf16 ----------------
__global__ void k_convert(const float* Wp, const float* Wr, const float* Wq,
                          const float* Wgate, const float* W_ih, const float* W_hh,
                          __bf16* wprt, __bf16* wqt, __bf16* wgt, __bf16* wihh) {
  int tid = blockIdx.x * blockDim.x + threadIdx.x;
  int nth = gridDim.x * blockDim.x;
  for (int i = tid; i < 512 * 1024; i += nth) {          // WprT[n][k] = k<512 ? Wp[k][n] : Wr[k-512][n]
    int n = i >> 10, k = i & 1023;
    float v = (k < 512) ? Wp[k * H_ + n] : Wr[(k - 512) * H_ + n];
    wprt[i] = (__bf16)v;
  }
  for (int i = tid; i < 512 * 512; i += nth) {           // WqT[n][k] = Wq[k][n]
    int n = i >> 9, k = i & 511;
    wqt[i] = (__bf16)Wq[k * H_ + n];
  }
  for (int i = tid; i < 1024 * 1024; i += nth) {         // WgateT[n][k] = Wgate[k][n]
    int n = i >> 10, k = i & 1023;
    wgt[i] = (__bf16)Wgate[k * R_ + n];
  }
  for (int i = tid; i < 2048 * 1536; i += nth) {         // Wihh[n][k] = k<1024 ? W_ih[n][k] : W_hh[n][k-1024]
    int n = i / 1536, k = i - n * 1536;
    float v = (k < 1024) ? W_ih[(size_t)n * R_ + k] : W_hh[(size_t)n * H_ + (k - 1024)];
    wihh[i] = (__bf16)v;
  }
}

// ---------------- one-time: zero state, barrier, mask bias ----------------
__global__ void k_init(float* h0, float* h1, float* c, float* negp, unsigned* bar, const int* mask) {
  int tid = blockIdx.x * blockDim.x + threadIdx.x;
  int nth = gridDim.x * blockDim.x;
  if (tid == 0) { bar[0] = 0u; bar[1] = 0u; }
  for (int i = tid; i < B_ * H_; i += nth) { h0[i] = 0.f; h1[i] = 0.f; c[i] = 0.f; }
  for (int i = tid; i < B_ * Q_; i += nth) negp[i] = mask[i] ? 0.f : -1e9f;
}

// ---------------- one-time GEMM: wq_hq = Hq @ Wq + bq  (M=16384, N=512, K=512) ----------------
__global__ void k_wqh(const float* Hq, const __bf16* wqt, const float* bq, float* wqh) {
  int lane = threadIdx.x & 31;
  int wv = threadIdx.x >> 5;
  int w = blockIdx.x * 8 + wv;                // one 16x16 tile per wave, 32768 tiles
  int trow = w >> 5, tcol = w & 31;
  int row0 = trow * 16, col0 = tcol * 16;
  v8f acc = {};
  for (int k0 = 0; k0 < DQ_; k0 += 32) {
    v16bf a = load_a_frag2(Hq, DQ_, 1 << 30, Hq, DQ_, row0, k0, lane);
    v16bf b = load_b_frag(wqt, DQ_, col0, k0, lane);
    acc = wmma_bf16(a, b, acc);
  }
  int n = col0 + (lane & 15);
  int hi = lane >> 4;
  float bn = bq[n];
#pragma unroll
  for (int v = 0; v < 8; ++v)
    wqh[(size_t)(row0 + v + 8 * hi) * H_ + n] = acc[v] + bn;
}

// ---------------- persistent timestep loop ----------------
__global__ void __launch_bounds__(NT, 1) k_rnn(
    const float* Hp, const float* Hq,
    const float* bp, const float* br,
    const float* wg, const float* bg, const float* bgate,
    const float* ln_g, const float* ln_b,
    const float* b_ih, const float* b_hh,
    const __bf16* wprt, const __bf16* wgt, const __bf16* wihh,
    const float* wqh, const float* negp,
    float* h0, float* h1, float* c, float* pre, float* qa, float* zi,
    float* out, unsigned* bar) {
  __shared__ float s_logits[2][Q_];
  __shared__ float s_alpha[2][Q_];
  int lane = threadIdx.x & 31;
  int wv = threadIdx.x >> 5;
  int w = blockIdx.x * 8 + wv;                 // global wave id, 0..511
  int hi = lane >> 4;
  int nlo = lane & 15;

  for (int t = 0; t < P_; ++t) {
    const float* hpi  = Hp + (size_t)t * DP_;  // row b lives at stride P_*DP_
    const float* hcur = (t & 1) ? h1 : h0;
    float*       hnxt = (t & 1) ? h0 : h1;

    // ---- Phase A: pre = hpi@Wp + hcur@Wr + bp + br   (M=128,N=512,K=1024 fused)
    if (w < 256) {
      int trow = w >> 5, tcol = w & 31;
      int row0 = trow * 16, col0 = tcol * 16;
      v8f acc = {};
      for (int k0 = 0; k0 < 1024; k0 += 32) {
        v16bf a = load_a_frag2(hpi, P_ * DP_, 512, hcur, H_, row0, k0, lane);
        v16bf b = load_b_frag(wprt, 1024, col0, k0, lane);
        acc = wmma_bf16(a, b, acc);
      }
      int n = col0 + nlo;
      float bb = bp[n] + br[n];
#pragma unroll
      for (int v = 0; v < 8; ++v)
        pre[(size_t)(row0 + v + 8 * hi) * H_ + n] = acc[v] + bb;
    } else if (t + 1 < P_) {
      // prefetch next timestep's Hp slice into cache (global_prefetch_b8)
      int e = (w - 256) * 32 + lane;           // 0..8191
      int r = e >> 6;                          // 0..127
      int col = (e & 63) * 8;                  // 0..504
      __builtin_prefetch(Hp + ((size_t)r * P_ + (t + 1)) * DP_ + col, 0, 3);
    }
    grid_barrier(bar);

    // ---- Phase B: logits -> softmax -> qa (4 waves per batch row; block-local sync)
    {
      int b = blockIdx.x * 2 + (wv >> 2);      // 0..127
      int bl = wv >> 2;                        // local row in LDS
      int qc = wv & 3;                         // q-chunk of 32
      const float* wqh_b = wqh + (size_t)b * Q_ * H_;
      const float* pre_b = pre + (size_t)b * H_;
      float4 pr4[4], wg4[4];
#pragma unroll
      for (int jj = 0; jj < 4; ++jj) {
        pr4[jj] = *(const float4*)(pre_b + jj * 128 + lane * 4);
        wg4[jj] = *(const float4*)(wg + jj * 128 + lane * 4);
      }
      float bgv = bg[0];
      for (int q = qc * 32; q < qc * 32 + 32; ++q) {
        const float* rowp = wqh_b + (size_t)q * H_;
        float s = 0.f;
#pragma unroll
        for (int jj = 0; jj < 4; ++jj) {
          float4 xv = *(const float4*)(rowp + jj * 128 + lane * 4);
          s += fast_tanh(xv.x + pr4[jj].x) * wg4[jj].x;
          s += fast_tanh(xv.y + pr4[jj].y) * wg4[jj].y;
          s += fast_tanh(xv.z + pr4[jj].z) * wg4[jj].z;
          s += fast_tanh(xv.w + pr4[jj].w) * wg4[jj].w;
        }
#pragma unroll
        for (int off = 16; off; off >>= 1) s += __shfl_xor(s, off, 32);
        if (lane == 0) s_logits[bl][q] = s + bgv + negp[b * Q_ + q];
      }
      __syncthreads();
      if (qc == 0) {                           // softmax over Q=128, one wave per row
        float lv[4], mx = -3.0e38f;
#pragma unroll
        for (int j = 0; j < 4; ++j) { lv[j] = s_logits[bl][lane + 32 * j]; mx = fmaxf(mx, lv[j]); }
#pragma unroll
        for (int off = 16; off; off >>= 1) mx = fmaxf(mx, __shfl_xor(mx, off, 32));
        float ss = 0.f;
#pragma unroll
        for (int j = 0; j < 4; ++j) { lv[j] = __expf(lv[j] - mx); ss += lv[j]; }
#pragma unroll
        for (int off = 16; off; off >>= 1) ss += __shfl_xor(ss, off, 32);
        float inv = __builtin_amdgcn_rcpf(ss);
#pragma unroll
        for (int j = 0; j < 4; ++j) s_alpha[bl][lane + 32 * j] = lv[j] * inv;
      }
      __syncthreads();
      // qa[b][d] = sum_q alpha[b][q] * Hq[b][q][d] ; 4 waves x 128 cols, float4 per lane
      const float* hq_b = Hq + (size_t)b * Q_ * DQ_;
      int d = qc * 128 + lane * 4;
      float4 a4 = make_float4(0.f, 0.f, 0.f, 0.f);
      for (int q = 0; q < Q_; ++q) {
        float av = s_alpha[bl][q];
        float4 hv = *(const float4*)(hq_b + (size_t)q * DQ_ + d);
        a4.x += av * hv.x; a4.y += av * hv.y; a4.z += av * hv.z; a4.w += av * hv.w;
      }
      *(float4*)(qa + (size_t)b * DQ_ + d) = a4;
    }
    grid_barrier(bar);

    // ---- Phase C: gate = sigmoid([hpi|qa]@Wgate + bgate); zi = gate*[hpi|qa]  (M=128,N=1024,K=1024)
    {
      int trow = w >> 6, tcol = w & 63;
      int row0 = trow * 16, col0 = tcol * 16;
      v8f acc = {};
      for (int k0 = 0; k0 < R_; k0 += 32) {
        v16bf a = load_a_frag2(hpi, P_ * DP_, 512, qa, DQ_, row0, k0, lane);
        v16bf b = load_b_frag(wgt, R_, col0, k0, lane);
        acc = wmma_bf16(a, b, acc);
      }
      int n = col0 + nlo;
      float bgt = bgate[n];
#pragma unroll
      for (int v = 0; v < 8; ++v) {
        int r = row0 + v + 8 * hi;
        float zr = (n < 512) ? hpi[(size_t)r * (P_ * DP_) + n]
                             : qa[(size_t)r * DQ_ + (n - 512)];
        zi[(size_t)r * R_ + n] = sigm(acc[v] + bgt) * zr;
      }
    }
    grid_barrier(bar);

    // ---- LayerNorm over R (in place; one wave per row)
    if (w < B_) {
      float* zb = zi + (size_t)w * R_;
      float zv[32];
      float s = 0.f;
#pragma unroll
      for (int j = 0; j < 32; ++j) { zv[j] = zb[lane + 32 * j]; s += zv[j]; }
#pragma unroll
      for (int off = 16; off; off >>= 1) s += __shfl_xor(s, off, 32);
      float mu = s * (1.0f / R_);
      float vv = 0.f;
#pragma unroll
      for (int j = 0; j < 32; ++j) { float d = zv[j] - mu; vv += d * d; }
#pragma unroll
      for (int off = 16; off; off >>= 1) vv += __shfl_xor(vv, off, 32);
      float rstd = rsqrtf(vv * (1.0f / R_) + LN_EPS);
#pragma unroll
      for (int j = 0; j < 32; ++j) {
        int n = lane + 32 * j;
        zb[n] = (zv[j] - mu) * rstd * ln_g[n] + ln_b[n];
      }
    }
    grid_barrier(bar);

    // ---- Phase D+E fused: gates = [zi|hcur]@[W_ih|W_hh]^T + b, then LSTM cell in-register.
    // One wave computes i,f,g,o quarters for the same (row,col) block: 4 accumulators, shared A.
    if (w < 256) {
      int trow = w >> 5, tcol = w & 31;        // 8 x 32 groups over (M=128, Hcols=512)
      int row0 = trow * 16, col0 = tcol * 16;
      v8f ai = {}, af = {}, ag = {}, ao = {};
      for (int k0 = 0; k0 < KD_; k0 += 32) {
        v16bf a = load_a_frag2(zi, R_, 1024, hcur, H_, row0, k0, lane);
        ai = wmma_bf16(a, load_b_frag(wihh, KD_, col0,        k0, lane), ai);
        af = wmma_bf16(a, load_b_frag(wihh, KD_, col0 + 512,  k0, lane), af);
        ag = wmma_bf16(a, load_b_frag(wihh, KD_, col0 + 1024, k0, lane), ag);
        ao = wmma_bf16(a, load_b_frag(wihh, KD_, col0 + 1536, k0, lane), ao);
      }
      int n = col0 + nlo;
      float bi_ = b_ih[n]        + b_hh[n];
      float bf_ = b_ih[n + 512]  + b_hh[n + 512];
      float bg_ = b_ih[n + 1024] + b_hh[n + 1024];
      float bo_ = b_ih[n + 1536] + b_hh[n + 1536];
#pragma unroll
      for (int v = 0; v < 8; ++v) {
        int r = row0 + v + 8 * hi;
        size_t idx = (size_t)r * H_ + n;
        float cv = c[idx];
        float iv = sigm(ai[v] + bi_);
        float fv = sigm(af[v] + bf_);
        float gv = fast_tanh(ag[v] + bg_);
        float ov = sigm(ao[v] + bo_);
        float cn = fv * cv + iv * gv;
        float hn = ov * fast_tanh(cn);
        c[idx] = cn;
        hnxt[idx] = hn;
        out[((size_t)r * P_ + t) * H_ + n] = hn;
      }
    }
    grid_barrier(bar);
  }
}

extern "C" void kernel_launch(void* const* d_in, const int* in_sizes, int n_in,
                              void* d_out, int out_size, void* d_ws, size_t ws_size,
                              hipStream_t stream) {
  const float* Hp    = (const float*)d_in[0];
  const float* Hq    = (const float*)d_in[1];
  const int*   mask  = (const int*)d_in[2];
  const float* Wq    = (const float*)d_in[3];
  const float* bq    = (const float*)d_in[4];
  const float* Wp    = (const float*)d_in[5];
  const float* bp    = (const float*)d_in[6];
  const float* Wr    = (const float*)d_in[7];
  const float* br    = (const float*)d_in[8];
  const float* wg    = (const float*)d_in[9];
  const float* bg    = (const float*)d_in[10];
  const float* Wgate = (const float*)d_in[11];
  const float* bgate = (const float*)d_in[12];
  const float* ln_g  = (const float*)d_in[13];
  const float* ln_b  = (const float*)d_in[14];
  const float* W_ih  = (const float*)d_in[15];
  const float* W_hh  = (const float*)d_in[16];
  const float* b_ih  = (const float*)d_in[17];
  const float* b_hh  = (const float*)d_in[18];
  float* out = (float*)d_out;

  char* ws = (char*)d_ws;
  unsigned* bar  = (unsigned*)(ws + OFF_BAR);
  float*  h0   = (float*)(ws + OFF_H0);
  float*  h1   = (float*)(ws + OFF_H1);
  float*  c    = (float*)(ws + OFF_C);
  float*  pre  = (float*)(ws + OFF_PRE);
  float*  qa   = (float*)(ws + OFF_QA);
  float*  zi   = (float*)(ws + OFF_ZI);
  float*  negp = (float*)(ws + OFF_NEG);
  float*  wqh  = (float*)(ws + OFF_WQH);
  __bf16* wprt = (__bf16*)(ws + OFF_WPR);
  __bf16* wqt  = (__bf16*)(ws + OFF_WQT);
  __bf16* wgt  = (__bf16*)(ws + OFF_WGT);
  __bf16* wihh = (__bf16*)(ws + OFF_WIH);

  k_convert<<<256, NT, 0, stream>>>(Wp, Wr, Wq, Wgate, W_ih, W_hh, wprt, wqt, wgt, wihh);
  k_init<<<64, NT, 0, stream>>>(h0, h1, c, negp, bar, mask);
  k_wqh<<<4096, NT, 0, stream>>>(Hq, wqt, bq, wqh);
  k_rnn<<<NWG, NT, 0, stream>>>(Hp, Hq, bp, br, wg, bg, bgate, ln_g, ln_b, b_ih, b_hh,
                                wprt, wgt, wihh, wqh, negp,
                                h0, h1, c, pre, qa, zi, out, bar);
}
